// InductiveGNN_79336635892669
// MI455X (gfx1250) — matrix operand order repaired and checked by reading, compiled, not run on hardware
//
#include <hip/hip_runtime.h>
#include <hip/hip_bf16.h>
#include <stdint.h>

typedef __attribute__((ext_vector_type(16))) _Float16 v16h;
typedef __attribute__((ext_vector_type(8)))  _Float16 v8h;
typedef __attribute__((ext_vector_type(8)))  float    v8f;

#define NN 50000
#define NE 1600000

static __device__ __forceinline__ float lrelu02(float v){ return v > 0.f ? v : 0.2f*v; }

// ---------------- utility kernels ----------------
__global__ void k_zero_f32(float* __restrict__ p, long n){
  long i = (long)blockIdx.x*blockDim.x + threadIdx.x;
  if (i < n) p[i] = 0.f;
}
__global__ void k_zero_u32(unsigned* __restrict__ p, long n){
  long i = (long)blockIdx.x*blockDim.x + threadIdx.x;
  if (i < n) p[i] = 0u;
}
__global__ void k_f2h(const float* __restrict__ s, _Float16* __restrict__ d, long n){
  long i = (long)blockIdx.x*blockDim.x + threadIdx.x;
  if (i < n) d[i] = (_Float16)s[i];
}
// W [H][K][F] f32 -> Wt [H][F][K] f16 (so GEMM B fragments are contiguous along K)
__global__ void k_wt(const float* __restrict__ W, _Float16* __restrict__ Wt, int H, int K, int F){
  long i = (long)blockIdx.x*blockDim.x + threadIdx.x;
  long tot = (long)H*K*F;
  if (i >= tot) return;
  int h = (int)(i / ((long)K*F));
  long r = i - (long)h*K*F;
  int k = (int)(r / F);
  int f = (int)(r - (long)k*F);
  Wt[((long)h*F + f)*K + k] = (_Float16)W[i];
}

// ---------------- WMMA GEMM ----------------
// C[h] = A(MxK) * B[h](KxF). One wave per (head, 16-row strip): loads the A
// fragment once per K-step and issues NT WMMAs against NT resident B fragments
// (NT = F/16 column tiles), maximizing wmma per byte fetched.
template<int NT>
__global__ void k_gemm_wmma(const _Float16* __restrict__ A, const _Float16* __restrict__ Bt,
                            float* __restrict__ C, int M, int K, int F, int H){
  int mt = M >> 4;
  long wg = ((long)blockIdx.x*blockDim.x + threadIdx.x) >> 5;
  long T = (long)H * mt;
  if (wg >= T) return;
  int lane = threadIdx.x & 31;
  int h  = (int)(wg / mt);
  int tm = (int)(wg - (long)h*mt);

  int m    = (tm << 4) + (lane & 15);        // A-fragment row
  int ncol = (lane & 15);                    // column within each 16-wide tile
  const _Float16* Arow  = A + (long)m*K;
  const _Float16* Bbase = Bt + ((long)h*F + ncol)*K;
  const int kbA = (lane < 16) ? 0 : 16;      // 16-bit A 16x32 layout, contiguous k = 16h + s (matches B)
  const int kbB = (lane < 16) ? 0 : 16;      // 16-bit B 32x16 layout

  v8f acc[NT];
  #pragma unroll
  for (int t = 0; t < NT; ++t) acc[t] = (v8f){};

  for (int k0 = 0; k0 < K; k0 += 32){
    v8h lo = *(const v8h*)(Arow + k0 + kbA);
    v8h hi = *(const v8h*)(Arow + k0 + kbA + 8);
    v16h a;
    #pragma unroll
    for (int j = 0; j < 8; ++j){ a[j] = lo[j]; a[8+j] = hi[j]; }
    #pragma unroll
    for (int t = 0; t < NT; ++t){
      v16h b = *(const v16h*)(Bbase + (long)(t*16)*K + k0 + kbB);
      acc[t] = __builtin_amdgcn_wmma_f32_16x16x32_f16(false, a, false, b, (short)0, acc[t],
                                                      false, false);
    }
  }
  // C layout [H][M][F]; lane holds column (t*16+ncol), rows m0..m0+7
  float* Cb = C + (long)h*M*F;
  int m0 = (tm << 4) + ((lane >> 4) << 3);
  #pragma unroll
  for (int t = 0; t < NT; ++t){
    #pragma unroll
    for (int q = 0; q < 8; ++q) Cb[(long)(m0+q)*F + t*16 + ncol] = acc[t][q];
  }
}

// ---------------- attention scalars: sd/ss [H][N] ----------------
__global__ void k_attn_scalars(const float* __restrict__ Wh, const float* __restrict__ adv,
                               const float* __restrict__ asv, float* __restrict__ sd,
                               float* __restrict__ ss, int Nn, int F, int H){
  long i = (long)blockIdx.x*blockDim.x + threadIdx.x;
  if (i >= (long)H*Nn) return;
  int h = (int)(i / Nn);
  const float* w  = Wh + i*F;
  const float* ad = adv + (long)h*F;
  const float* as = asv + (long)h*F;
  float d = 0.f, s = 0.f;
  for (int f = 0; f < F; ++f){ float v = w[f]; d += v*ad[f]; s += v*as[f]; }
  sd[i] = d; ss[i] = s;
}

// ---------------- edge passes ----------------
static __device__ __forceinline__ unsigned f32_key(float x){
  unsigned b = __float_as_uint(x);
  return (b & 0x80000000u) ? ~b : (b | 0x80000000u);   // order-preserving map
}
__global__ void k_edge_max(const int* __restrict__ ei, const float* __restrict__ sd,
                           const float* __restrict__ ss, unsigned* __restrict__ mkey,
                           int E, int Nn, int H){
  long e = (long)blockIdx.x*blockDim.x + threadIdx.x;
  if (e >= E) return;
  int s = ei[e], d = ei[(long)E + e];
  for (int h = 0; h < H; ++h){
    float v = lrelu02(sd[(long)h*Nn + d] + ss[(long)h*Nn + s]);
    atomicMax(&mkey[(long)h*Nn + d], f32_key(v));
  }
}
__global__ void k_max_fin(const unsigned* __restrict__ mkey, float* __restrict__ m, long n){
  long i = (long)blockIdx.x*blockDim.x + threadIdx.x;
  if (i >= n) return;
  unsigned k = mkey[i];
  float v = 0.f;                           // matches reference: non-finite max -> 0
  if (k != 0u){
    unsigned b = (k & 0x80000000u) ? (k & 0x7fffffffu) : ~k;
    v = __uint_as_float(b);
  }
  m[i] = v;
}
__global__ void k_edge_sum(const int* __restrict__ ei, const float* __restrict__ sd,
                           const float* __restrict__ ss, const float* __restrict__ m,
                           float* __restrict__ ssum, int E, int Nn, int H){
  long e = (long)blockIdx.x*blockDim.x + threadIdx.x;
  if (e >= E) return;
  int s = ei[e], d = ei[(long)E + e];
  for (int h = 0; h < H; ++h){
    float v = lrelu02(sd[(long)h*Nn + d] + ss[(long)h*Nn + s]);
    atomicAdd(&ssum[(long)h*Nn + d], __expf(v - m[(long)h*Nn + d]));
  }
}
// one wave32 per (edge, head); lanes stripe over features (coalesced gather + atomic scatter)
__global__ void k_edge_agg(const int* __restrict__ ei, const float* __restrict__ Wh,
                           const float* __restrict__ sd, const float* __restrict__ ss,
                           const float* __restrict__ m, const float* __restrict__ ssum,
                           float* __restrict__ hout, int E, int Nn, int F, int H){
  long wid = ((long)blockIdx.x*blockDim.x + threadIdx.x) >> 5;
  if (wid >= (long)E*H) return;
  int lane = threadIdx.x & 31;
  int h = (int)(wid % H);
  long e = wid / H;
  int s = ei[e], d = ei[(long)E + e];
  const float* wsrc = Wh + ((long)h*Nn + s)*F;
  __builtin_prefetch(wsrc + lane, 0, 1);   // global_prefetch: hide random-gather latency
  float v = lrelu02(sd[(long)h*Nn + d] + ss[(long)h*Nn + s]);
  float alpha = __expf(v - m[(long)h*Nn + d]) / (ssum[(long)h*Nn + d] + 1e-16f);
  float* outr = hout + (long)d*((long)H*F) + (long)h*F;
  for (int f = lane; f < F; f += 32)
    atomicAdd(&outr[f], alpha * wsrc[f]);
}

// ---------------- epilogues ----------------
__global__ void k_elu(float* __restrict__ p, long n){
  long i = (long)blockIdx.x*blockDim.x + threadIdx.x;
  if (i >= n) return;
  float x = p[i];
  p[i] = x > 0.f ? x : (__expf(x) - 1.f);
}
__global__ void k_elu_add(float* __restrict__ p, const float* __restrict__ r, long n){
  long i = (long)blockIdx.x*blockDim.x + threadIdx.x;
  if (i >= n) return;
  float x = p[i];
  x = x > 0.f ? x : (__expf(x) - 1.f);
  p[i] = x + r[i];
}
__global__ void k_out(const float* __restrict__ h3, float* __restrict__ out, int n){
  int i = blockIdx.x*blockDim.x + threadIdx.x;
  if (i >= n) return;
  const float* row = h3 + (long)i*96;
  float v[16];
  #pragma unroll
  for (int f = 0; f < 16; ++f){
    float acc = 0.f;
    #pragma unroll
    for (int h = 0; h < 6; ++h){
      float x = row[h*16 + f];
      x = x > 0.f ? x : (__expf(x) - 1.f);
      acc += x;
    }
    v[f] = acc * (1.f/6.f);
  }
  float mx = v[0];
  #pragma unroll
  for (int f = 1; f < 16; ++f) mx = fmaxf(mx, v[f]);
  float sum = 0.f;
  #pragma unroll
  for (int f = 0; f < 16; ++f){ v[f] = __expf(v[f] - mx); sum += v[f]; }
  float inv = 1.f / sum;
  #pragma unroll
  for (int f = 0; f < 16; ++f) out[(long)i*16 + f] = v[f]*inv;
}

// ---------------- host orchestration ----------------
namespace {
struct Scratch {
  _Float16* xh; _Float16* wt;
  float* Wh; float* sd; float* ss;
  unsigned* mkey; float* m; float* ssum;
  float* h1; float* h2; float* h3;
};
inline unsigned nblk(long n, int b){ return (unsigned)((n + b - 1)/b); }

void run_gat_layer(hipStream_t st, const float* hin, int Fin, int Fout, int H,
                   const float* W, const float* ad, const float* as,
                   const int* ei, float* hout, const Scratch& w){
  const int B = 256;
  long nc = (long)NN*Fin;
  k_f2h<<<nblk(nc,B), B, 0, st>>>(hin, w.xh, nc);
  long nw = (long)H*Fin*Fout;
  k_wt<<<nblk(nw,B), B, 0, st>>>(W, w.wt, H, Fin, Fout);
  long T = (long)H * (NN/16);               // one wave per (head, 16-row strip)
  if (Fout == 64)
    k_gemm_wmma<4><<<nblk(T*32,B), B, 0, st>>>(w.xh, w.wt, w.Wh, NN, Fin, Fout, H);
  else
    k_gemm_wmma<1><<<nblk(T*32,B), B, 0, st>>>(w.xh, w.wt, w.Wh, NN, Fin, Fout, H);
  long ns = (long)H*NN;
  k_attn_scalars<<<nblk(ns,B), B, 0, st>>>(w.Wh, ad, as, w.sd, w.ss, NN, Fout, H);
  k_zero_u32<<<nblk(ns,B), B, 0, st>>>(w.mkey, ns);
  k_zero_f32<<<nblk(ns,B), B, 0, st>>>(w.ssum, ns);
  long no = (long)NN*H*Fout;
  k_zero_f32<<<nblk(no,B), B, 0, st>>>(hout, no);
  k_edge_max<<<nblk(NE,B), B, 0, st>>>(ei, w.sd, w.ss, w.mkey, NE, NN, H);
  k_max_fin<<<nblk(ns,B), B, 0, st>>>(w.mkey, w.m, ns);
  k_edge_sum<<<nblk(NE,B), B, 0, st>>>(ei, w.sd, w.ss, w.m, w.ssum, NE, NN, H);
  long aw = (long)NE*H*32;
  k_edge_agg<<<nblk(aw,B), B, 0, st>>>(ei, w.Wh, w.sd, w.ss, w.m, w.ssum, hout, NE, NN, Fout, H);
}
} // namespace

extern "C" void kernel_launch(void* const* d_in, const int* in_sizes, int n_in,
                              void* d_out, int out_size, void* d_ws, size_t ws_size,
                              hipStream_t stream){
  const float* x   = (const float*)d_in[0];
  const int*   ei  = (const int*)  d_in[1];
  const float* W1  = (const float*)d_in[2];
  const float* a1d = (const float*)d_in[3];
  const float* a1s = (const float*)d_in[4];
  const float* W2  = (const float*)d_in[5];
  const float* a2d = (const float*)d_in[6];
  const float* a2s = (const float*)d_in[7];
  const float* W3  = (const float*)d_in[8];
  const float* a3d = (const float*)d_in[9];
  const float* a3s = (const float*)d_in[10];

  char* base = (char*)d_ws;
  size_t off = 0;
  auto take = [&](size_t bytes)->char*{
    char* p = base + off;
    off += (bytes + 255) & ~(size_t)255;
    return p;
  };
  Scratch w;
  w.xh   = (_Float16*)take((size_t)NN*256*2);    // f16 layer input, max Fin=256
  w.wt   = (_Float16*)take((size_t)6*64*256*2);  // transposed f16 weights (max over layers)
  w.Wh   = (float*)   take((size_t)4*NN*64*4);   // per-head projected features (max)
  w.sd   = (float*)   take((size_t)6*NN*4);
  w.ss   = (float*)   take((size_t)6*NN*4);
  w.mkey = (unsigned*)take((size_t)6*NN*4);
  w.m    = (float*)   take((size_t)6*NN*4);
  w.ssum = (float*)   take((size_t)6*NN*4);
  w.h1   = (float*)   take((size_t)NN*256*4);
  w.h2   = (float*)   take((size_t)NN*256*4);
  w.h3   = (float*)   take((size_t)NN*96*4);

  const int B = 256;
  // ---- layer 1: 128 -> 4x64, ELU+concat ----
  run_gat_layer(stream, x, 128, 64, 4, W1, a1d, a1s, ei, w.h1, w);
  long n1 = (long)NN*256;
  k_elu<<<nblk(n1,B), B, 0, stream>>>(w.h1, n1);
  // ---- layer 2: 256 -> 4x64, ELU+concat, residual with h1 ----
  run_gat_layer(stream, w.h1, 256, 64, 4, W2, a2d, a2s, ei, w.h2, w);
  k_elu_add<<<nblk(n1,B), B, 0, stream>>>(w.h2, w.h1, n1);
  // ---- layer 3: 256 -> 6x16, ELU -> mean heads -> softmax ----
  run_gat_layer(stream, w.h2, 256, 16, 6, W3, a3d, a3s, ei, w.h3, w);
  k_out<<<nblk(NN,B), B, 0, stream>>>(w.h3, (float*)d_out, NN);
}
